// CrossAttention_80642305950080
// MI455X (gfx1250) — matrix operand order repaired and checked by reading, not compiled, not run on hardware
//
#include <hip/hip_runtime.h>
#include <hip/hip_bf16.h>
#include <math.h>

// ---- shapes (compile-time constants for this problem) ----
#define BB   4
#define NN   1024
#define PP   1024
#define CC   512
#define HH   8
#define HC   4096   // H*C

typedef __attribute__((ext_vector_type(16))) __bf16 v16bf;
typedef __attribute__((ext_vector_type(8)))  __bf16 v8bf;
typedef __attribute__((ext_vector_type(8)))  float  v8f;
typedef __attribute__((ext_vector_type(4)))  unsigned int v4u;
typedef __attribute__((ext_vector_type(8)))  int v8i;
typedef __attribute__((ext_vector_type(4)))  int v4i;

__device__ __forceinline__ unsigned short f2bf(float f) {
    unsigned int u = __float_as_uint(f);
    u += 0x7FFFu + ((u >> 16) & 1u);   // round-to-nearest-even
    return (unsigned short)(u >> 16);
}

__device__ __forceinline__ v16bf cat8(v8bf lo, v8bf hi) {
    return __builtin_shufflevector(lo, hi, 0,1,2,3,4,5,6,7,8,9,10,11,12,13,14,15);
}

// A-matrix 16x32 bf16 fragment (CDNA5 wave32 layout):
// lane L: row = L&15 ; kbase = (L<16)?0:8
// elements 0..7  -> K = kbase + 0..7 ; elements 8..15 -> K = 16 + kbase + 0..7
__device__ __forceinline__ v16bf load_frag_a(const unsigned short* base, int rowStride,
                                             int lane, int colBase) {
    int row = lane & 15;
    int kb  = (lane < 16) ? 0 : 8;
    const unsigned short* p = base + (long)row * rowStride + colBase + kb;
    v8bf lo = *(const v8bf*)p;
    v8bf hi = *(const v8bf*)(p + 16);
    return cat8(lo, hi);
}

// B-matrix 32x16 bf16 fragment: lane L: n = L&15 ; koff = (L<16)?0:8
// elements 0..7 -> K = koff + 0..7 ; elements 8..15 -> K = 16 + koff + 0..7. Source laid out n-major with K contiguous.
__device__ __forceinline__ v16bf load_frag_b(const unsigned short* base, int nStride,
                                             int lane, int nBase, int kBase) {
    int n    = lane & 15;
    int koff = (lane < 16) ? 0 : 8;
    const unsigned short* p = base + (long)(nBase + n) * nStride + kBase + koff;
    v8bf lo = *(const v8bf*)p;
    v8bf hi = *(const v8bf*)(p + 16);
    return cat8(lo, hi);
}

__device__ __forceinline__ v8f wmma_bf16(v16bf a, v16bf b, v8f c) {
    return __builtin_amdgcn_wmma_f32_16x16x32_bf16(false, a, false, b, (short)0, c,
                                                   false, false);
}

// ---- Tensor Data Mover: 2D tile load (bf16 elements) into LDS ----
// D# per CDNA5 ISA ch.8: group0 = {count/flags, lds_addr, global_addr lo,
// global_addr hi | type=2}; group1 carries data_size=1 (2B), tensor dims =
// tile dims (tiles are always in-bounds here), dim0 stride, optional LDS pad.
// This toolchain exposes the 6-arg builtin:
//   (uint32x4 g0, int32x8 g1, int32x4 g2, int32x4 g3, int32x8 extra, i32 cpol)
__device__ __forceinline__ void tdm_load_2d(unsigned lds_addr, const void* gptr,
                                            unsigned tile_d0, unsigned tile_d1,
                                            unsigned long stride0_elems,
                                            int pad_en, unsigned pad_int_code,
                                            unsigned pad_amt_code) {
    unsigned long ga = (unsigned long)gptr;
    v4u g0 = { 1u,                                   // count=1, user descriptor
               lds_addr,                             // LDS byte address
               (unsigned)(ga & 0xFFFFFFFFu),
               (unsigned)((ga >> 32) & 0x01FFFFFFu) | 0x80000000u };  // type=2
    int w0 = 0x00010000                              // data_size = 1 (2 bytes)
           | (pad_en ? (1 << 20) : 0)
           | (int)(pad_int_code << 22)
           | (int)(pad_amt_code << 25);
    v8i g1 = { w0,
               (int)((tile_d0 & 0xFFFFu) << 16),                                  // tensor_dim0 lo
               (int)(((tile_d0 >> 16) & 0xFFFFu) | ((tile_d1 & 0xFFFFu) << 16)),  // dim0 hi | dim1 lo
               (int)(((tile_d1 >> 16) & 0xFFFFu) | ((tile_d0 & 0xFFFFu) << 16)),  // dim1 hi | tile_dim0
               (int)(tile_d1 & 0xFFFFu),                                          // tile_dim1 (tile_dim2=0)
               (int)(stride0_elems & 0xFFFFFFFFu),                                // dim0 stride lo
               (int)((stride0_elems >> 32) & 0xFFFFu),                            // dim0 stride hi
               0 };
    v4i gz4 = { 0, 0, 0, 0 };
    v8i gz8 = { 0, 0, 0, 0, 0, 0, 0, 0 };
    __builtin_amdgcn_tensor_load_to_lds(g0, g1, gz4, gz4, gz8, 0);
}

__device__ __forceinline__ unsigned lds_addr_of(const void* p) {
    return (unsigned)(unsigned long long)p;   // LDS aperture: addr[31:0] = LDS offset
}

// -------------------- fp32 -> bf16 converter --------------------
__global__ __launch_bounds__(256)
void f32_to_bf16_kernel(const float* __restrict__ in, unsigned short* __restrict__ out, long n) {
    long i = (long)blockIdx.x * blockDim.x + threadIdx.x;
    long stride = (long)gridDim.x * blockDim.x;
    for (; i < n; i += stride) out[i] = f2bf(in[i]);
}

// -------------------- LayerNorm (C=512) -> bf16 --------------------
__global__ __launch_bounds__(256)
void ln_bf16_kernel(const float* __restrict__ x, const float* __restrict__ g,
                    const float* __restrict__ bt, unsigned short* __restrict__ out) {
    __shared__ float s1[256];
    __shared__ float s2[256];
    int t = threadIdx.x;
    long row = blockIdx.x;
    const float* xr = x + row * CC;
    float a = xr[t], b = xr[t + 256];
    s1[t] = a + b;
    s2[t] = a * a + b * b;
    __syncthreads();
    for (int s = 128; s > 0; s >>= 1) {
        if (t < s) { s1[t] += s1[t + s]; s2[t] += s2[t + s]; }
        __syncthreads();
    }
    float mu   = s1[0] * (1.0f / CC);
    float var  = s2[0] * (1.0f / CC) - mu * mu;
    float rinv = rsqrtf(var + 1e-5f);
    out[row * CC + t]       = f2bf((a - mu) * rinv * g[t]       + bt[t]);
    out[row * CC + t + 256] = f2bf((b - mu) * rinv * g[t + 256] + bt[t + 256]);
}

// -------------------- generic batched bf16 WMMA GEMM --------------------
// D = scale * (A @ B) [+ Res] ; A: MxK (row major, lda), B: KxN (row major, ldb)
// Block tile 64x128, 8 waves; wave = 32x32 output (4 WMMAs/k-step, A+B frag reuse).
// A tile staged by the Tensor Data Mover with LDS padding (40-half pitch).
__global__ __launch_bounds__(256)
void gemm_bf16_kernel(const unsigned short* __restrict__ A,
                      const unsigned short* __restrict__ B,
                      unsigned short* __restrict__ Dbf,
                      float* __restrict__ Df,
                      const float* __restrict__ Res,
                      int M, int Nn, int Kd, int lda, int ldb, int ldd,
                      float scale, int Z0,
                      long sA0, long sA1, long sB0, long sB1, long sD0, long sD1) {
    __shared__ unsigned short As[64 * 40];    // 64 rows x 32 K (+8 pad)
    __shared__ unsigned short Bt[128 * 40];   // 128 n x 32 K (transposed), K contiguous

    int z = blockIdx.z, z0 = z % Z0, z1 = z / Z0;
    const unsigned short* Ap = A + z1 * sA1 + z0 * sA0;
    const unsigned short* Bp = B + z1 * sB1 + z0 * sB0;
    long dOff = z1 * sD1 + z0 * sD0;

    int n0 = blockIdx.x * 128, m0 = blockIdx.y * 64;
    int t = threadIdx.x, lane = t & 31, w = t >> 5;
    int wm = w & 1, wn = w >> 1;              // wave owns rows 32*wm.., cols 32*wn..
    unsigned as_lds = lds_addr_of(&As[0]);

    v8f acc00 = {}, acc01 = {}, acc10 = {}, acc11 = {};

    for (int k0 = 0; k0 < Kd; k0 += 32) {
        __syncthreads();
        {   // stage B tile (32 x 128), transposed so fragments read contiguous K
            int r = t >> 3, cs = (t & 7) * 16;
            const unsigned short* p = Bp + (long)(k0 + r) * ldb + n0 + cs;
            v8bf v0 = *(const v8bf*)p;
            v8bf v1 = *(const v8bf*)(p + 8);
            const unsigned short* vp0 = (const unsigned short*)&v0;
            const unsigned short* vp1 = (const unsigned short*)&v1;
            #pragma unroll
            for (int j = 0; j < 8; ++j) {
                Bt[(cs + j) * 40 + r]     = vp0[j];
                Bt[(cs + 8 + j) * 40 + r] = vp1[j];
            }
        }
        if (w == 0) {
            // TDM: A tile 64 rows x 32 elems, row stride lda; LDS pad 16B per 64B
            // (pad_interval code 3 = 16 DWORDs, pad_amount code 3 = 4 DWORDs)
            tdm_load_2d(as_lds, Ap + (long)m0 * lda + k0, 32, 64,
                        (unsigned long)lda, 1, 3, 3);
            __builtin_amdgcn_s_wait_tensorcnt(0);
        }
        if (k0 + 32 < Kd) {   // gfx1250 global_prefetch for next tiles
            __builtin_prefetch(Ap + (long)(m0 + (t >> 2)) * lda + k0 + 32, 0, 1);
            __builtin_prefetch(Bp + (long)(k0 + 32 + (t >> 3)) * ldb + n0, 0, 1);
        }
        __syncthreads();

        v16bf a0 = load_frag_a(As + (32 * wm) * 40,      40, lane, 0);
        v16bf a1 = load_frag_a(As + (32 * wm + 16) * 40, 40, lane, 0);
        v16bf b0 = load_frag_b(Bt, 40, lane, 32 * wn,      0);
        v16bf b1 = load_frag_b(Bt, 40, lane, 32 * wn + 16, 0);
        acc00 = wmma_bf16(a0, b0, acc00);
        acc01 = wmma_bf16(a0, b1, acc01);
        acc10 = wmma_bf16(a1, b0, acc10);
        acc11 = wmma_bf16(a1, b1, acc11);
    }

    // epilogue: C/D layout -> lane = N, vgpr(+8*hi) = M
    int hi = lane >> 4, nl = lane & 15;
    #pragma unroll
    for (int mb = 0; mb < 2; ++mb) {
        #pragma unroll
        for (int nb = 0; nb < 2; ++nb) {
            v8f c = mb ? (nb ? acc11 : acc10) : (nb ? acc01 : acc00);
            int colG = n0 + 32 * wn + 16 * nb + nl;
            #pragma unroll
            for (int i = 0; i < 8; ++i) {
                int rowG = m0 + 32 * wm + 16 * mb + i + 8 * hi;
                float v = c[i] * scale;
                if (Df) {
                    float r = Res ? Res[(long)rowG * ldd + colG] : 0.f;
                    Df[dOff + (long)rowG * ldd + colG] = v + r;
                } else {
                    Dbf[dOff + (long)rowG * ldd + colG] = f2bf(v);
                }
            }
        }
    }
}

// -------------------- fused flash attention + ELU --------------------
// grid = (N/16, B*H); block = 256 (8 waves). Wave w owns O columns [64w,64w+64).
// Q/K tiles staged via Tensor Data Mover; V tile manually transposed in LDS.
__global__ __launch_bounds__(256)
void attn_kernel(const unsigned short* __restrict__ Q,
                 const unsigned short* __restrict__ Kt,
                 const unsigned short* __restrict__ V,
                 unsigned short* __restrict__ E) {
    __shared__ unsigned short Qs[16 * 512];      // 16 KB
    __shared__ unsigned short U[512 * 40];       // 40 KB: K tile (32x512) / V^T tile
    __shared__ float S_lds[2 * 16 * 16];
    __shared__ unsigned short P_lds[16 * 32];
    __shared__ float m_lds[16], l_lds[16], alpha_lds[16];

    int t = threadIdx.x, lane = t & 31, w = t >> 5;
    int bh = blockIdx.y, b = bh >> 3, h = bh & 7;
    int n0 = blockIdx.x * 16;
    int cw = 64 * w;
    const float SCALE = 0.044194173824159216f;   // 1/sqrt(512)

    const unsigned short* Qh = Q  + (long)b * NN * HC + (long)h * CC;
    const unsigned short* Kh = Kt + (long)bh * PP * CC;
    const unsigned short* Vh = V  + (long)b * PP * HC + (long)h * CC;
    unsigned qs_lds = lds_addr_of(&Qs[0]);
    unsigned u_lds  = lds_addr_of(&U[0]);

    if (w == 0) {   // TDM: Q tile = 16 rows x 512 elems, row stride HC
        tdm_load_2d(qs_lds, Qh + (long)n0 * HC, 512, 16, HC, 0, 0, 0);
        __builtin_amdgcn_s_wait_tensorcnt(0);
    }
    if (t < 16) { m_lds[t] = -1e30f; l_lds[t] = 0.f; }

    v8f o0 = {}, o1 = {}, o2 = {}, o3 = {};

    for (int pt = 0; pt < PP; pt += 32) {
        __syncthreads();
        S_lds[t] = 0.f; S_lds[t + 256] = 0.f;
        if (w == 0) {   // TDM: contiguous 32x512 K tile (1D, 16384 elems)
            tdm_load_2d(u_lds, Kh + (long)pt * CC, 16384, 1, 16384, 0, 0, 0);
            __builtin_amdgcn_s_wait_tensorcnt(0);
        }
        __syncthreads();
        {   // partial scores over this wave's 64-channel slice
            v8f s0 = {}, s1 = {};
            #pragma unroll
            for (int kk = 0; kk < 2; ++kk) {
                int ccb = cw + 32 * kk;
                v16bf a  = load_frag_a(Qs, 512, lane, ccb);
                v16bf b0 = load_frag_b(U, 512, lane, 0,  ccb);
                v16bf b1 = load_frag_b(U, 512, lane, 16, ccb);
                s0 = wmma_bf16(a, b0, s0);
                s1 = wmma_bf16(a, b1, s1);
            }
            int hi = lane >> 4, nl = lane & 15;
            #pragma unroll
            for (int i = 0; i < 8; ++i) {
                int row = i + 8 * hi;
                atomicAdd(&S_lds[row * 16 + nl],       s0[i]);
                atomicAdd(&S_lds[256 + row * 16 + nl], s1[i]);
            }
        }
        __syncthreads();
        if (t < 16) {   // online softmax for row t
            float mold = m_lds[t], mnew = mold;
            #pragma unroll
            for (int j = 0; j < 32; ++j) {
                float s = S_lds[(j >> 4) * 256 + t * 16 + (j & 15)] * SCALE;
                mnew = fmaxf(mnew, s);
            }
            float alpha = __expf(mold - mnew), sum = 0.f;
            #pragma unroll
            for (int j = 0; j < 32; ++j) {
                float s = S_lds[(j >> 4) * 256 + t * 16 + (j & 15)] * SCALE;
                float p = __expf(s - mnew);
                sum += p;
                P_lds[t * 32 + j] = f2bf(p);
            }
            m_lds[t] = mnew;
            l_lds[t] = l_lds[t] * alpha + sum;
            alpha_lds[t] = alpha;
        }
        __syncthreads();
        {   // rescale running O by alpha(row)
            int hi = lane >> 4;
            #pragma unroll
            for (int i = 0; i < 8; ++i) {
                float al = alpha_lds[i + 8 * hi];
                o0[i] *= al; o1[i] *= al; o2[i] *= al; o3[i] *= al;
            }
        }
        {   // stage V tile transposed: U[col][key], pitch 40
            int r = t >> 3, cs = (t & 7) * 64;
            const unsigned short* p = Vh + (long)(pt + r) * HC + cs;
            #pragma unroll
            for (int i = 0; i < 8; ++i) {
                v8bf v = *(const v8bf*)(p + 8 * i);
                const unsigned short* vp = (const unsigned short*)&v;
                #pragma unroll
                for (int j = 0; j < 8; ++j)
                    U[(cs + 8 * i + j) * 40 + r] = vp[j];
            }
        }
        __syncthreads();
        {   // O += P~ (16x32) @ V_tile (32 x 64-slice)
            v16bf ap = load_frag_a(P_lds, 32, lane, 0);
            v16bf b0 = load_frag_b(U, 40, lane, cw + 0,  0);
            v16bf b1 = load_frag_b(U, 40, lane, cw + 16, 0);
            v16bf b2 = load_frag_b(U, 40, lane, cw + 32, 0);
            v16bf b3 = load_frag_b(U, 40, lane, cw + 48, 0);
            o0 = wmma_bf16(ap, b0, o0);
            o1 = wmma_bf16(ap, b1, o1);
            o2 = wmma_bf16(ap, b2, o2);
            o3 = wmma_bf16(ap, b3, o3);
        }
    }
    __syncthreads();
    {   // finalize: divide by row sum, fused ELU, store bf16
        int hi = lane >> 4, nl = lane & 15;
        #pragma unroll
        for (int i = 0; i < 8; ++i) {
            int row = i + 8 * hi;
            float linv = 1.f / l_lds[row];
            float vals[4] = { o0[i] * linv, o1[i] * linv, o2[i] * linv, o3[i] * linv };
            long rowOff = ((long)b * NN + n0 + row) * HC + (long)h * CC + cw + nl;
            #pragma unroll
            for (int f = 0; f < 4; ++f) {
                float x = vals[f];
                float e = x > 0.f ? x : (__expf(x) - 1.f);
                E[rowOff + 16 * f] = f2bf(e);
            }
        }
    }
}

// -------------------- host-side launch --------------------
extern "C" void kernel_launch(void* const* d_in, const int* in_sizes, int n_in,
                              void* d_out, int out_size, void* d_ws, size_t ws_size,
                              hipStream_t stream) {
    (void)in_sizes; (void)n_in; (void)out_size; (void)ws_size;
    const float* target = (const float*)d_in[0];
    const float* source = (const float*)d_in[1];
    const float* ln_g   = (const float*)d_in[2];
    const float* ln_b   = (const float*)d_in[3];
    const float* Wq     = (const float*)d_in[4];
    const float* Wk     = (const float*)d_in[5];
    const float* Wv     = (const float*)d_in[6];
    const float* Wa     = (const float*)d_in[7];
    const float* Wo     = (const float*)d_in[8];
    float* out = (float*)d_out;

    char* ws = (char*)d_ws;
    size_t off = 0;
    auto alloc = [&](size_t bytes) -> void* {
        void* p = ws + off;
        off = (off + bytes + 255) & ~(size_t)255;
        return p;
    };

    const long BN = (long)BB * NN;        // 4096
    const long BP = (long)BB * PP;        // 4096
    unsigned short* tbf  = (unsigned short*)alloc(BN * CC * 2);
    unsigned short* sbf  = (unsigned short*)alloc(BP * CC * 2);
    unsigned short* wqb  = (unsigned short*)alloc((long)CC * HC * 2);
    unsigned short* wkb  = (unsigned short*)alloc((long)CC * HC * 2);
    unsigned short* wvb  = (unsigned short*)alloc((long)CC * HC * 2);
    unsigned short* wab  = (unsigned short*)alloc((long)HH * CC * CC * 2);
    unsigned short* wob  = (unsigned short*)alloc((long)HC * CC * 2);
    unsigned short* Qb   = (unsigned short*)alloc(BN * HC * 2);
    unsigned short* K0b  = (unsigned short*)alloc(BP * HC * 2);
    unsigned short* Vb   = (unsigned short*)alloc(BP * HC * 2);
    unsigned short* Ktb  = (unsigned short*)alloc((long)BB * HH * PP * CC * 2);
    unsigned short* Eb   = (unsigned short*)alloc(BN * HC * 2);

    const float inv_sqrt_c = 0.044194173824159216f; // 1/sqrt(512)

    // 1) LayerNorm -> bf16 ; bf16 conversions
    ln_bf16_kernel<<<dim3((unsigned)BN), dim3(256), 0, stream>>>(target, ln_g, ln_b, tbf);
    f32_to_bf16_kernel<<<dim3(1024), dim3(256), 0, stream>>>(source, sbf, BP * CC);
    f32_to_bf16_kernel<<<dim3(1024), dim3(256), 0, stream>>>(Wq, wqb, (long)CC * HC);
    f32_to_bf16_kernel<<<dim3(1024), dim3(256), 0, stream>>>(Wk, wkb, (long)CC * HC);
    f32_to_bf16_kernel<<<dim3(1024), dim3(256), 0, stream>>>(Wv, wvb, (long)CC * HC);
    f32_to_bf16_kernel<<<dim3(1024), dim3(256), 0, stream>>>(Wa, wab, (long)HH * CC * CC);
    f32_to_bf16_kernel<<<dim3(1024), dim3(256), 0, stream>>>(Wo, wob, (long)HC * CC);

    // 2) Projections: Q = t@Wq, K0 = src@Wk, V = src@Wv   (4096x512 . 512x4096)
    gemm_bf16_kernel<<<dim3(HC / 128, BN / 64, 1), dim3(256), 0, stream>>>(
        tbf, wqb, Qb, nullptr, nullptr, (int)BN, HC, CC, CC, HC, HC, 1.f, 1,
        0, 0, 0, 0, 0, 0);
    gemm_bf16_kernel<<<dim3(HC / 128, BP / 64, 1), dim3(256), 0, stream>>>(
        sbf, wkb, K0b, nullptr, nullptr, (int)BP, HC, CC, CC, HC, HC, 1.f, 1,
        0, 0, 0, 0, 0, 0);
    gemm_bf16_kernel<<<dim3(HC / 128, BP / 64, 1), dim3(256), 0, stream>>>(
        sbf, wvb, Vb, nullptr, nullptr, (int)BP, HC, CC, CC, HC, HC, 1.f, 1,
        0, 0, 0, 0, 0, 0);

    // 3) Per-head bilinear K transform: Kt[b,h] = K0[b,:,h] @ W_att[h] * 1/sqrt(C)
    gemm_bf16_kernel<<<dim3(CC / 128, PP / 64, BB * HH), dim3(256), 0, stream>>>(
        K0b, wab, Ktb, nullptr, nullptr, PP, CC, CC, HC, CC, CC, inv_sqrt_c, HH,
        /*sA0=*/CC, /*sA1=*/(long)PP * HC,
        /*sB0=*/(long)CC * CC, /*sB1=*/0,
        /*sD0=*/(long)PP * CC, /*sD1=*/(long)HH * PP * CC);

    // 4) Fused flash attention + ELU -> E (B*N, HC) bf16
    attn_kernel<<<dim3(NN / 16, BB * HH), dim3(256), 0, stream>>>(Qb, Ktb, Vb, Eb);

    // 5) Output projection + residual: out = target + E @ Wo   (fp32)
    gemm_bf16_kernel<<<dim3(CC / 128, BN / 64, 1), dim3(256), 0, stream>>>(
        Eb, wob, nullptr, out, target, (int)BN, CC, HC, HC, CC, CC, 1.f, 1,
        0, 0, 0, 0, 0, 0);
}
